// VectorQuantize_84035330114069
// MI455X (gfx1250) — compile-verified, matrix operand order checked by reading, not run on hardware
//
#include <hip/hip_runtime.h>
#include <hip/hip_bf16.h>

typedef __attribute__((ext_vector_type(16))) _Float16 v16h;
typedef __attribute__((ext_vector_type(8)))  float    v8f;

#define BQ 8
#define NQ 16384
#define DQ 64
#define CQ 2048
#define ROWS (BQ * NQ)              /* 131072 */
#define BLOCK_THREADS 256
#define ROWS_PER_BLOCK 128          /* 8 waves * 16 rows */
#define CHUNK 256                   /* codes staged per LDS pass (32 KB) */
#define NCHUNK (CQ / CHUNK)         /* 8 */
#define TILES_PER_CHUNK (CHUNK / 16)

#define IDX_MASK 0x7FFu             /* 11 bits: C = 2048 columns */

// ---------------------------------------------------------------------------
// Kernel 1: L2-normalize codebook rows, fp32 -> f16, into workspace.
// ---------------------------------------------------------------------------
__global__ __launch_bounds__(32)
void vq_norm_embed(const float* __restrict__ embed, _Float16* __restrict__ enorm) {
    const int c    = blockIdx.x;
    const int lane = threadIdx.x;                 // 0..31
    const float2 f = ((const float2*)(embed + (size_t)c * DQ))[lane];
    float p = f.x * f.x + f.y * f.y;
    #pragma unroll
    for (int off = 16; off >= 1; off >>= 1) p += __shfl_xor(p, off, 32);
    const float s = 1.0f / fmaxf(sqrtf(p), 1e-12f);
    _Float16* o = enorm + (size_t)c * DQ + lane * 2;
    o[0] = (_Float16)(f.x * s);
    o[1] = (_Float16)(f.y * s);
}

// ---------------------------------------------------------------------------
// Kernel 2: WMMA similarity + packed-index argmax + gather.
// ---------------------------------------------------------------------------
__global__ __launch_bounds__(BLOCK_THREADS)
void vq_main(const float* __restrict__ x,
             const float* __restrict__ embed,
             const _Float16* __restrict__ enorm,
             float* __restrict__ outq,
             float* __restrict__ outi) {
    __shared__ uint4 lds[CHUNK * 8];              // 256 codes * 128 B = 32 KB

    const int tid   = threadIdx.x;
    const int lane  = tid & 31;
    const int wave  = tid >> 5;
    const int l15   = lane & 15;
    const int hi    = lane >> 4;                  // 0 or 1 (which 16-lane half)
    const int kbase = hi * 8;

    const int rowbase = blockIdx.x * ROWS_PER_BLOCK + wave * 16;
    const int row     = rowbase + l15;            // lane L and L+16 share a row

    // ---- Load A tile (16 rows x 64), normalize per row, convert to f16 ----
    // 16-bit A 16x32 layout: lane holds K = {kbase..kbase+7, 16+kbase..23+kbase}
    float f[32];
    {
        const float* xr = x + (size_t)row * DQ;
        #pragma unroll
        for (int ks = 0; ks < 2; ++ks) {
            const float* p0 = xr + ks * 32 + kbase;
            #pragma unroll
            for (int i = 0; i < 8; ++i) f[ks * 16 + i]     = p0[i];
            #pragma unroll
            for (int i = 0; i < 8; ++i) f[ks * 16 + 8 + i] = p0[16 + i];
        }
    }
    float p = 0.0f;
    #pragma unroll
    for (int i = 0; i < 32; ++i) p += f[i] * f[i];
    p += __shfl_xor(p, 16, 32);                   // other half of this row
    const float s = 1.0f / fmaxf(sqrtf(p), 1e-12f);

    v16h a0, a1;
    #pragma unroll
    for (int i = 0; i < 16; ++i) {
        a0[i] = (_Float16)(f[i]      * s);
        a1[i] = (_Float16)(f[16 + i] * s);
    }

    // Packed (score | encoded-column) running maxima. Low 11 mantissa bits of
    // the f32 score carry (col ^ 0x7FF): plain float max does argmax with
    // lowest-index tie preference (for positive scores). Init = large negative
    // finite (never -inf: OR-ing index bits into -inf would make a NaN).
    float best[8];
    #pragma unroll
    for (int q = 0; q < 8; ++q) best[q] = __uint_as_float(0xFF000000u);

    // Per-lane part of the encoded column (bits 0..3), tile part is bits 4..10.
    const unsigned encv = (unsigned)(l15 ^ 0xF);

    // ---- Sweep the codebook in 32 KB LDS chunks ----
    for (int cc = 0; cc < NCHUNK; ++cc) {
        __syncthreads();                          // previous chunk fully consumed
        {
            const int code = cc * CHUNK + tid;    // 256 threads <-> 256 codes
            const uint4* g = (const uint4*)(enorm + (size_t)code * DQ);
            uint4* lr = lds + tid * 8;
            const int sw = tid & 7;
            #pragma unroll
            for (int j = 0; j < 8; ++j) lr[j ^ sw] = g[j];   // XOR-swizzled
            if (cc + 1 < NCHUNK)
                __builtin_prefetch(enorm + (size_t)(code + CHUNK) * DQ, 0, 1);
        }
        __syncthreads();

        #pragma unroll 2
        for (int tp = 0; tp < TILES_PER_CHUNK; tp += 2) {
            v8f c0 = {}, c1 = {};
            #pragma unroll
            for (int u = 0; u < 2; ++u) {
                const int nloc = (tp + u) * 16 + l15;   // lane = column N of B
                const uint4* br = lds + nloc * 8;
                const int sw = nloc & 7;
                union { v16h v; uint4 w[2]; } b0, b1;
                b0.w[0] = br[(0 + hi) ^ sw];      // K =  kbase..kbase+7
                b0.w[1] = br[(2 + hi) ^ sw];      // K = 16+kbase..
                b1.w[0] = br[(4 + hi) ^ sw];      // K = 32+kbase..
                b1.w[1] = br[(6 + hi) ^ sw];      // K = 48+kbase..
                v8f c = {};
                c = __builtin_amdgcn_wmma_f32_16x16x32_f16(false, a0, false, b0.v,
                                                           (short)0, c, false, false);
                c = __builtin_amdgcn_wmma_f32_16x16x32_f16(false, a1, false, b1.v,
                                                           (short)0, c, false, false);
                if (u == 0) c0 = c; else c1 = c;
            }

            // Hoisted encoded columns: one v_or each, pinned to a single VGPR
            // so the per-score pack stays a fused (and,or) op.
            unsigned enc0 = ((unsigned)((cc * CHUNK + (tp + 0) * 16) ^ 0x7F0)) | encv;
            unsigned enc1 = ((unsigned)((cc * CHUNK + (tp + 1) * 16) ^ 0x7F0)) | encv;
            asm("" : "+v"(enc0));
            asm("" : "+v"(enc1));

            #pragma unroll
            for (int q = 0; q < 8; ++q) {
                const unsigned u0 = (__float_as_uint(c0[q]) & ~IDX_MASK) | enc0;
                const unsigned u1 = (__float_as_uint(c1[q]) & ~IDX_MASK) | enc1;
                // v_and_or_b32 x2 + v_max3_num_f32
                best[q] = fmaxf(best[q],
                                fmaxf(__uint_as_float(u0), __uint_as_float(u1)));
            }
        }
    }

    // ---- Argmax reduction across the 16 lanes sharing each row ----
    #pragma unroll
    for (int off = 8; off >= 1; off >>= 1) {
        #pragma unroll
        for (int q = 0; q < 8; ++q)
            best[q] = fmaxf(best[q], __shfl_xor(best[q], off, 32));
    }
    // lane 0 now holds rows rowbase+0..7; lane 16 rows rowbase+8..15

    // ---- Gather un-normalized codebook rows + emit indices ----
    #pragma unroll
    for (int j = 0; j < 16; ++j) {
        const float bv = __shfl(best[j & 7], (j & 8) ? 16 : 0, 32);
        const int idx  = (int)((__float_as_uint(bv) & IDX_MASK) ^ IDX_MASK);
        const float2 v = ((const float2*)(embed + (size_t)idx * DQ))[lane];
        ((float2*)(outq + (size_t)(rowbase + j) * DQ))[lane] = v;
        if (lane == 0) outi[rowbase + j] = (float)idx;
    }
}

// ---------------------------------------------------------------------------
extern "C" void kernel_launch(void* const* d_in, const int* in_sizes, int n_in,
                              void* d_out, int out_size, void* d_ws, size_t ws_size,
                              hipStream_t stream) {
    const float* x     = (const float*)d_in[0];   // [8,16384,64] fp32
    const float* embed = (const float*)d_in[1];   // [1,2048,64]  fp32
    _Float16*    enorm = (_Float16*)d_ws;         // 2048*64 f16 = 256 KB
    float* outq = (float*)d_out;                  // quantize [8,16384,64]
    float* outi = outq + (size_t)ROWS * DQ;       // indices  [8,16384]

    vq_norm_embed<<<CQ, 32, 0, stream>>>(embed, enorm);
    vq_main<<<ROWS / ROWS_PER_BLOCK, BLOCK_THREADS, 0, stream>>>(
        x, embed, enorm, outq, outi);
}
